// MultiHeadAttention_77240691851739
// MI455X (gfx1250) — compile-verified, matrix operand order checked by reading, not run on hardware
//
#include <hip/hip_runtime.h>

// Problem constants (from reference): B=2, S=2048, H=1024, NH=16, HD=64
#define Bq  2
#define Sq  2048
#define Hq  1024
#define NHq 16
#define HDq 64
// All GEMMs are M=4096, N=1024, K=1024 (compile-time -> cheap addressing)
#define GN  1024
#define GK  1024

typedef __attribute__((ext_vector_type(16))) __bf16 v16bf;
typedef __attribute__((ext_vector_type(2)))  __bf16 v2bf;
typedef __attribute__((ext_vector_type(8)))  float  v8f;

// f32 pair -> packed bf16 (v_cvt_pk_bf16_f32)
__device__ __forceinline__ unsigned int pk_bf16(float x, float y) {
  v2bf t;
  t[0] = (__bf16)x;
  t[1] = (__bf16)y;
  return __builtin_bit_cast(unsigned int, t);
}
__device__ __forceinline__ unsigned short bf16u(float f) {
  return __builtin_bit_cast(unsigned short, (__bf16)f);
}

// Build a 16xbf16 WMMA fragment from two contiguous 16B runs
// (fragment elements 0..7 = first K-run, 8..15 = second K-run).
struct U32x8 { uint4 lo, hi; };
__device__ __forceinline__ v16bf ldfrag(const unsigned short* p_lo,
                                        const unsigned short* p_hi) {
  U32x8 r;
  r.lo = *(const uint4*)p_lo;   // ds_load_b128 / global_load_b128
  r.hi = *(const uint4*)p_hi;
  return __builtin_bit_cast(v16bf, r);
}

// LDS offset (addrspace(3) byte offset) of a __shared__ object
__device__ __forceinline__ unsigned int lds_off(const void* p) {
  return (unsigned int)(unsigned long long)
      (__attribute__((address_space(3))) const void*)p;
}

// Max-reduce across the 16 lanes of a half-wave group using DPP16 (VALU pipe,
// keeps the DS pipe free for WMMA fragment traffic).
__device__ __forceinline__ float dpp_max16(float x) {
  int t;
  t = __builtin_amdgcn_update_dpp(0, __float_as_int(x), 0xB1,  0xF, 0xF, true); // quad_perm [1,0,3,2]
  x = fmaxf(x, __int_as_float(t));
  t = __builtin_amdgcn_update_dpp(0, __float_as_int(x), 0x4E,  0xF, 0xF, true); // quad_perm [2,3,0,1]
  x = fmaxf(x, __int_as_float(t));
  t = __builtin_amdgcn_update_dpp(0, __float_as_int(x), 0x141, 0xF, 0xF, true); // row_half_mirror
  x = fmaxf(x, __int_as_float(t));
  t = __builtin_amdgcn_update_dpp(0, __float_as_int(x), 0x140, 0xF, 0xF, true); // row_mirror
  x = fmaxf(x, __int_as_float(t));
  return x;
}

// ---------------------------------------------------------------------------
// GEMM: out = A(f32, Mx1024) @ W(f32, 1024x1024) + bias, bf16 WMMA, f32 accum.
// MODE 0: write bf16, head-split layout [B, NH, S, HD]   (Q/K/V projections)
// MODE 1: write f32,  flat [M, N]                         (output projection)
// 128 threads (4 waves), 64x64 tile, 32x32 per wave, K-step 32,
// double-buffered LDS: global loads of tile t+1 overlap WMMAs of tile t.
// ---------------------------------------------------------------------------
template <int MODE>
__global__ __launch_bounds__(128) void mha_gemm_bias(
    const float* __restrict__ A, const float* __restrict__ W,
    const float* __restrict__ bias, void* __restrict__ outp) {
  __shared__ __align__(16) unsigned short Asm[2][64][32];  // [buf][m][k]
  __shared__ __align__(16) unsigned short Bs[2][64][32];   // [buf][n][k] (transposed)

  const int row0 = blockIdx.y * 64;
  const int col0 = blockIdx.x * 64;
  const int tid  = threadIdx.x;
  const int lane = tid & 31, wave = tid >> 5;
  const int l16  = lane & 15, half = lane >> 4;
  const int wr   = (wave & 1) * 32;
  const int wc   = (wave >> 1) * 32;

  // Staging coordinates (fixed per thread)
  const int a_r  = tid >> 3;          // A row base (+i*16)
  const int a_c  = (tid & 7) * 4;     // A k-column
  const int b_kp = tid >> 4;          // B k-pair base (+i*8)
  const int b_n  = (tid & 15) * 4;    // B n-column base
  const float* Ap = A + (size_t)(row0 + a_r) * GK + a_c;
  const float* Wp = W + (size_t)(b_kp * 2) * GN + col0 + b_n;

  float4 ta[4], wa[2], wb[2];
  auto load_tile = [&](int kk) {
#pragma unroll
    for (int i = 0; i < 4; ++i)
      ta[i] = *(const float4*)(Ap + kk + i * 16 * GK);
#pragma unroll
    for (int i = 0; i < 2; ++i) {
      wa[i] = *(const float4*)(Wp + kk * GN + i * 16 * GN);
      wb[i] = *(const float4*)(Wp + kk * GN + i * 16 * GN + GN);
    }
  };
  auto store_tile = [&](int nb) {
#pragma unroll
    for (int i = 0; i < 4; ++i)
      *(uint2*)&Asm[nb][a_r + i * 16][a_c] =
          make_uint2(pk_bf16(ta[i].x, ta[i].y), pk_bf16(ta[i].z, ta[i].w));
#pragma unroll
    for (int i = 0; i < 2; ++i) {
      int kp2 = (b_kp + i * 8) * 2;
      *(unsigned int*)&Bs[nb][b_n + 0][kp2] = pk_bf16(wa[i].x, wb[i].x);
      *(unsigned int*)&Bs[nb][b_n + 1][kp2] = pk_bf16(wa[i].y, wb[i].y);
      *(unsigned int*)&Bs[nb][b_n + 2][kp2] = pk_bf16(wa[i].z, wb[i].z);
      *(unsigned int*)&Bs[nb][b_n + 3][kp2] = pk_bf16(wa[i].w, wb[i].w);
    }
  };

  v8f z = {};
  v8f acc[2][2] = {{z, z}, {z, z}};

  load_tile(0);
  store_tile(0);
  int bufi = 0;
  for (int kk = 0; kk < GK; kk += 32) {
    __syncthreads();  // buf writes visible; previous buf readers done
    if (kk + 32 < GK) load_tile(kk + 32);  // overlaps WMMAs below

    v16bf af[2], bfr[2];
#pragma unroll
    for (int mt = 0; mt < 2; ++mt) {
      const unsigned short* ar = Asm[bufi][wr + mt * 16 + l16];
      af[mt] = ldfrag(&ar[half * 8], &ar[16 + half * 8]);
    }
#pragma unroll
    for (int nt = 0; nt < 2; ++nt) {
      const unsigned short* br = Bs[bufi][wc + nt * 16 + l16];
      bfr[nt] = ldfrag(&br[half * 8], &br[16 + half * 8]);
    }
#pragma unroll
    for (int mt = 0; mt < 2; ++mt)
#pragma unroll
      for (int nt = 0; nt < 2; ++nt)
        acc[mt][nt] = __builtin_amdgcn_wmma_f32_16x16x32_bf16(
            false, af[mt], false, bfr[nt], (short)0, acc[mt][nt], false, false);

    if (kk + 32 < GK) store_tile(bufi ^ 1);
    bufi ^= 1;
  }

  // Epilogue: bias + store. C layout: col = l16, row = r + 8*half.
#pragma unroll
  for (int mt = 0; mt < 2; ++mt)
#pragma unroll
    for (int nt = 0; nt < 2; ++nt) {
      int n = col0 + wc + nt * 16 + l16;
      float bv = bias[n];
#pragma unroll
      for (int r = 0; r < 8; ++r) {
        int mrow = row0 + wr + mt * 16 + r + 8 * half;
        float v = acc[mt][nt][r] + bv;
        if (MODE == 0) {
          int bb = mrow / Sq, ss = mrow % Sq;
          int hh = n / HDq,  dd = n % HDq;
          ((unsigned short*)outp)[(((size_t)bb * NHq + hh) * Sq + ss) * HDq + dd] =
              bf16u(v);
        } else {
          ((float*)outp)[(size_t)mrow * GN + n] = v;
        }
      }
    }
}

// ---------------------------------------------------------------------------
// Flash attention: per (b,h), 64 query rows per workgroup (16 per wave),
// K/V streamed in blocks of 32 with DOUBLE-BUFFERED LDS:
//   K block:  GLOBAL_LOAD_ASYNC_TO_LDS_B128 (ASYNCcnt DMA, overlaps compute)
//   V block:  register 2x2 bf16 transpose -> Vs[d][key]
// Online softmax: row-max via DPP16 (VALU), row-sum via P@ones WMMA.
// Output: merged heads f32 [B, S, H].
// ---------------------------------------------------------------------------
__global__ __launch_bounds__(128) void mha_flash_attn(
    const unsigned short* __restrict__ Qh, const unsigned short* __restrict__ Kh,
    const unsigned short* __restrict__ Vh, const float* __restrict__ mask,
    float* __restrict__ out) {
  __shared__ __align__(16) unsigned short Ksm[2][32][64];  // [buf][key][d]
  __shared__ __align__(16) unsigned short Vs[2][64][32];   // [buf][d][key]
  __shared__ __align__(16) unsigned short Psm[4][16][32];  // per-wave scratch

  const int tid  = threadIdx.x;
  const int lane = tid & 31, wave = tid >> 5;
  const int l16  = lane & 15, half = lane >> 4;
  const int q0   = blockIdx.x * 64;
  const int bh   = blockIdx.y;
  const int bb   = bh / NHq, hh = bh % NHq;

  // Q fragments (A layout): two global_load_b128 per d-half
  v16bf qf[2];
  {
    const unsigned short* qrow =
        Qh + ((size_t)bh * Sq + q0 + wave * 16 + l16) * HDq;
#pragma unroll
    for (int dh = 0; dh < 2; ++dh)
      qf[dh] = ldfrag(&qrow[dh * 32 + half * 8], &qrow[dh * 32 + 16 + half * 8]);
  }
  // All-ones B fragment: rowsum(P) = P @ ones (lands in C layout, every
  // column equal -> per-lane element r == l for row r+8*half).
  v16bf ones;
#pragma unroll
  for (int e = 0; e < 16; ++e) ones[e] = (__bf16)1.0f;

  // Issue K (async DMA to LDS) + V (register transpose) for block j0n -> buf nb
  auto issue_kv = [&](int j0n, int nb) {
    const unsigned short* kb = Kh + ((size_t)bh * Sq + j0n) * HDq;
    const unsigned short* vb = Vh + ((size_t)bh * Sq + j0n) * HDq;
    unsigned long long kbase = (unsigned long long)kb;
    unsigned int lbase = lds_off(&Ksm[nb][0][0]);
#pragma unroll
    for (int i = 0; i < 2; ++i) {
      unsigned int goff  = (unsigned int)((tid + i * 128) * 16);
      unsigned int laddr = lbase + goff;
      asm volatile("global_load_async_to_lds_b128 %0, %1, %2 offset:0"
                   :: "v"(laddr), "v"(goff), "s"(kbase) : "memory");
    }
    // V: keys kp*2, kp*2+1; d-range dq*8..dq*8+7; 2x b128 loads, 8 b32 stores
    int kp = tid >> 3, dq = tid & 7;
    uint4 a = *(const uint4*)&vb[(size_t)(kp * 2    ) * 64 + dq * 8];
    uint4 b = *(const uint4*)&vb[(size_t)(kp * 2 + 1) * 64 + dq * 8];
    unsigned int au[4] = {a.x, a.y, a.z, a.w};
    unsigned int bu[4] = {b.x, b.y, b.z, b.w};
#pragma unroll
    for (int j = 0; j < 4; ++j) {
      *(unsigned int*)&Vs[nb][dq * 8 + 2 * j    ][kp * 2] =
          (au[j] & 0xFFFFu) | (bu[j] << 16);
      *(unsigned int*)&Vs[nb][dq * 8 + 2 * j + 1][kp * 2] =
          (au[j] >> 16) | (bu[j] & 0xFFFF0000u);
    }
  };

  v8f z = {};
  v8f oacc[4] = {z, z, z, z};
  v8f lacc = z;      // row-sum accumulator (5th WMMA tile)
  float m_r[8];
#pragma unroll
  for (int r = 0; r < 8; ++r) m_r[r] = -1e30f;

  const float* mask_b = mask + (size_t)bb * Sq * Sq;

  issue_kv(0, 0);
  int bufi = 0;
  for (int j0 = 0; j0 < Sq; j0 += 32) {
    asm volatile("s_wait_asynccnt 0x0" ::: "memory");  // own K DMA done
    __syncthreads();                                   // all waves' buf ready
    if (j0 + 32 < Sq) issue_kv(j0 + 32, bufi ^ 1);     // overlaps compute below

    // Scores: two 16x16 tiles (columns j0+cb*16..+15), contraction over HD=64
    v8f sc[2] = {z, z};
#pragma unroll
    for (int cb = 0; cb < 2; ++cb)
#pragma unroll
      for (int dh = 0; dh < 2; ++dh) {
        const unsigned short* kr = Ksm[bufi][cb * 16 + l16];
        v16bf kf = ldfrag(&kr[dh * 32 + half * 8], &kr[dh * 32 + 16 + half * 8]);
        sc[cb] = __builtin_amdgcn_wmma_f32_16x16x32_bf16(
            false, qf[dh], false, kf, (short)0, sc[cb], false, false);
      }

    // scale 1/sqrt(64) + mask, then online softmax (max via DPP16)
    float p[2][8], resc[8];
#pragma unroll
    for (int r = 0; r < 8; ++r) {
      int q = q0 + wave * 16 + r + 8 * half;
      const float* mr = mask_b + (size_t)q * Sq + j0;
      float s0 = sc[0][r] * 0.125f + mr[l16];
      float s1 = sc[1][r] * 0.125f + mr[16 + l16];
      p[0][r] = s0; p[1][r] = s1;
      float tmax = dpp_max16(fmaxf(s0, s1));       // row max over 32 columns
      float mnew = fmaxf(m_r[r], tmax);
      resc[r] = __expf(m_r[r] - mnew);
      m_r[r] = mnew;
      p[0][r] = __expf(s0 - mnew);
      p[1][r] = __expf(s1 - mnew);
    }
    // rescale running accumulators (O and l)
#pragma unroll
    for (int nt = 0; nt < 4; ++nt)
#pragma unroll
      for (int r = 0; r < 8; ++r) oacc[nt][r] *= resc[r];
#pragma unroll
    for (int r = 0; r < 8; ++r) lacc[r] *= resc[r];

    // P: C-layout -> A-layout transpose through per-wave LDS scratch
#pragma unroll
    for (int cb = 0; cb < 2; ++cb)
#pragma unroll
      for (int r = 0; r < 8; ++r)
        Psm[wave][r + 8 * half][cb * 16 + l16] = bf16u(p[cb][r]);
    asm volatile("s_wait_dscnt 0" ::: "memory");  // wave-local RAW on LDS

    const unsigned short* pr = Psm[wave][l16];
    v16bf pf = ldfrag(&pr[half * 8], &pr[16 + half * 8]);

    // l += rowsum(P) on the matrix pipe
    lacc = __builtin_amdgcn_wmma_f32_16x16x32_bf16(
        false, pf, false, ones, (short)0, lacc, false, false);

    // O += P @ V   (contraction over the 32 keys; 4 d-tiles of 16)
#pragma unroll
    for (int nt = 0; nt < 4; ++nt) {
      const unsigned short* vr = Vs[bufi][nt * 16 + l16];
      v16bf vf = ldfrag(&vr[half * 8], &vr[16 + half * 8]);
      oacc[nt] = __builtin_amdgcn_wmma_f32_16x16x32_bf16(
          false, pf, false, vf, (short)0, oacc[nt], false, false);
    }
    bufi ^= 1;
  }

  // epilogue: normalize and write merged-head layout [B, S, H]
#pragma unroll
  for (int nt = 0; nt < 4; ++nt)
#pragma unroll
    for (int r = 0; r < 8; ++r) {
      int q = q0 + wave * 16 + r + 8 * half;
      int d = nt * 16 + l16;
      out[((size_t)bb * Sq + q) * Hq + hh * HDq + d] = oacc[nt][r] / lacc[r];
    }
}

// ---------------------------------------------------------------------------
extern "C" void kernel_launch(void* const* d_in, const int* in_sizes, int n_in,
                              void* d_out, int out_size, void* d_ws, size_t ws_size,
                              hipStream_t stream) {
  const float* query = (const float*)d_in[0];
  const float* key_  = (const float*)d_in[1];
  const float* value = (const float*)d_in[2];
  const float* amask = (const float*)d_in[3];
  const float* Wq = (const float*)d_in[4];  const float* bq = (const float*)d_in[5];
  const float* Wk = (const float*)d_in[6];  const float* bk = (const float*)d_in[7];
  const float* Wv = (const float*)d_in[8];  const float* bv = (const float*)d_in[9];
  const float* Wo = (const float*)d_in[10]; const float* bo = (const float*)d_in[11];

  const int M = Bq * Sq;  // 4096

  // Workspace: 3 x bf16 QKV (head-split) + f32 attention output = 40 MB
  char* ws = (char*)d_ws;
  size_t bf16_buf = (size_t)M * Hq * sizeof(unsigned short);  // 8 MB each
  unsigned short* Qh = (unsigned short*)(ws);
  unsigned short* Kh = (unsigned short*)(ws + bf16_buf);
  unsigned short* Vh = (unsigned short*)(ws + 2 * bf16_buf);
  float*          at = (float*)(ws + 3 * bf16_buf);           // 16 MB

  dim3 gg(GN / 64, M / 64);   // 16 x 64 blocks, 128 threads each
  mha_gemm_bias<0><<<gg, 128, 0, stream>>>(query, Wq, bq, Qh);
  mha_gemm_bias<0><<<gg, 128, 0, stream>>>(key_,  Wk, bk, Kh);
  mha_gemm_bias<0><<<gg, 128, 0, stream>>>(value, Wv, bv, Vh);

  dim3 ga(Sq / 64, Bq * NHq);  // 32 x 32 blocks
  mha_flash_attn<<<ga, 128, 0, stream>>>(Qh, Kh, Vh, amask, at);

  mha_gemm_bias<1><<<gg, 128, 0, stream>>>(at, Wo, bo, d_out);
}
